// Transformer_14688788152787
// MI455X (gfx1250) — matrix operand order repaired and checked by reading, not compiled, not run on hardware
//
#include <hip/hip_runtime.h>
#include <stdint.h>

// ======================================================================
// Complex transformer decoder layer for MI455X (gfx1250, wave32, WMMA).
// All GEMMs run on v_wmma_f32_16x16x32_f16 with fp32 accumulation.
// Complex GEMM = 4 real WMMA accumulators combined at the epilogue.
// GEMM tiles are staged into LDS with global_load_async_to_lds_b128
// (double buffered, s_wait_asynccnt). Attention keeps a full 64x1024
// fp32 logits tile in LDS (263KB of the 320KB WGP LDS) so softmax is a
// simple in-LDS pass, then P*V uses WMMA against a pre-transposed V.
// Requires d_ws >= ~510MB.
// ======================================================================

#define EPSF 1e-5f
#define LDT 40   // LDS tile row stride in halfs (keeps 16B align, staggers banks)

typedef __attribute__((ext_vector_type(16))) _Float16 v16h;
typedef __attribute__((ext_vector_type(8)))  _Float16 v8h;
typedef __attribute__((ext_vector_type(8)))  float    v8f;

union HF16 { v16h v; v8h h[2]; };

__device__ __forceinline__ v8f wmma_f16(v16h a, v16h b, v8f c) {
  // (neg_a, A, neg_b, B, c_mod, C, reuse_a, reuse_b)
  return __builtin_amdgcn_wmma_f32_16x16x32_f16(false, a, false, b, (short)0, c, false, false);
}

// A-fragment (16x32 f16): lane holds row (lane&15); halfs [ka..ka+8) and
// [ka+16..ka+24) with ka=(lane>>4)*8. Caller passes p = &row[ka].
__device__ __forceinline__ v16h frag_a(const _Float16* p) {
  HF16 u;
  u.h[0] = *(const v8h*)(p);
  u.h[1] = *(const v8h*)(p + 16);
  return u.v;
}
// B-fragment (32x16 f16): lane holds column (lane&15); halfs [kb..kb+8) and
// [kb+16..kb+24) with kb=(lane>>4)*8 (same k order as frag_a). The tile is
// stored N-major (W rows), caller passes p = &colrow[kb].
__device__ __forceinline__ v16h frag_b(const _Float16* p) {
  HF16 u;
  u.h[0] = *(const v8h*)(p);
  u.h[1] = *(const v8h*)(p + 16);
  return u.v;
}

__device__ __forceinline__ void async_wait0() {
  asm volatile("s_wait_asynccnt 0" ::: "memory");
}

// Copy a 64-row x 32-half tile global -> LDS via async DMA (no VGPR staging).
// 256 16B chunks; 128 threads issue 2 each (one lane per chunk).
__device__ __forceinline__ void async_tile_64x32(_Float16* dst, const _Float16* src,
                                                 int ld, int tid) {
  for (int h = 0; h < 2; ++h) {
    int c   = tid + h * 128;          // 0..255
    int row = c >> 2;                 // 4 chunks per row
    int kc  = (c & 3) << 3;           // half offset within row
    unsigned lds  = (unsigned)(uintptr_t)(dst + row * LDT + kc);
    unsigned goff = ((unsigned)row * (unsigned)ld + (unsigned)kc) * 2u;
    asm volatile("global_load_async_to_lds_b128 %0, %1, %2"
                 :: "v"(lds), "v"(goff), "s"((unsigned long long)(uintptr_t)src)
                 : "memory");
  }
}

// ---------------------------------------------------------------------------
// Complex GEMM: C(MxN) = A(MxK) * W(NxK)^T + bias.  A/W as split f16 planes.
// Block = 64x64 complex C, 4 waves, wave = 32x32 (2x2 of 16x16), K step 32.
// Outputs: optional fp32 planes and/or f16 planes (optionally transposed).
// ---------------------------------------------------------------------------
__global__ __launch_bounds__(128) void cgemm_kernel(
    const _Float16* __restrict__ Ar, const _Float16* __restrict__ Ai, int lda,
    const _Float16* __restrict__ Wr, const _Float16* __restrict__ Wi, int ldw,
    const float2* __restrict__ bias,
    float* __restrict__ Cr32, float* __restrict__ Ci32, int ldc32,
    _Float16* __restrict__ Cr16, _Float16* __restrict__ Ci16, long long ldc16,
    int K, int trans, int relu)
{
  __shared__ _Float16 sA[2][2][64 * LDT];   // [buf][plane]
  __shared__ _Float16 sW[2][2][64 * LDT];

  const int tid  = threadIdx.x;
  const int lane = tid & 31, wave = tid >> 5;
  const int row0 = blockIdx.y * 64, col0 = blockIdx.x * 64;
  const int wr = (wave >> 1) * 32, wc = (wave & 1) * 32;

  const _Float16* gA[2] = { Ar + (size_t)row0 * lda, Ai + (size_t)row0 * lda };
  const _Float16* gW[2] = { Wr + (size_t)col0 * ldw, Wi + (size_t)col0 * ldw };

  const v8f zero8 = {0.f,0.f,0.f,0.f,0.f,0.f,0.f,0.f};
  v8f arr[2][2], aii[2][2], ari[2][2], air[2][2];
  for (int i = 0; i < 2; ++i)
    for (int j = 0; j < 2; ++j) { arr[i][j]=zero8; aii[i][j]=zero8; ari[i][j]=zero8; air[i][j]=zero8; }

  const int nk = K >> 5;
  for (int p = 0; p < 2; ++p) {
    async_tile_64x32(&sA[0][p][0], gA[p], lda, tid);
    async_tile_64x32(&sW[0][p][0], gW[p], ldw, tid);
  }
  async_wait0();
  __syncthreads();

  const int frow = lane & 15;
  const int ka = (lane >> 4) * 8;
  const int kb = (lane >> 4) * 8;

  for (int kt = 0; kt < nk; ++kt) {
    const int cur = kt & 1;
    if (kt + 1 < nk) {
      const int off = (kt + 1) << 5;
      for (int p = 0; p < 2; ++p) {
        async_tile_64x32(&sA[cur ^ 1][p][0], gA[p] + off, lda, tid);
        async_tile_64x32(&sW[cur ^ 1][p][0], gW[p] + off, ldw, tid);
      }
    }
    v16h aR[2], aI[2], bR[2], bI[2];
#pragma unroll
    for (int i = 0; i < 2; ++i) {
      const int r = (wr + 16 * i + frow) * LDT;
      aR[i] = frag_a(&sA[cur][0][r + ka]);
      aI[i] = frag_a(&sA[cur][1][r + ka]);
    }
#pragma unroll
    for (int j = 0; j < 2; ++j) {
      const int r = (wc + 16 * j + frow) * LDT;
      bR[j] = frag_b(&sW[cur][0][r + kb]);
      bI[j] = frag_b(&sW[cur][1][r + kb]);
    }
#pragma unroll
    for (int i = 0; i < 2; ++i)
#pragma unroll
      for (int j = 0; j < 2; ++j) {
        arr[i][j] = wmma_f16(aR[i], bR[j], arr[i][j]);   // Re += Ar*Wr
        aii[i][j] = wmma_f16(aI[i], bI[j], aii[i][j]);   // Re -= Ai*Wi (at epilogue)
        ari[i][j] = wmma_f16(aR[i], bI[j], ari[i][j]);   // Im += Ar*Wi
        air[i][j] = wmma_f16(aI[i], bR[j], air[i][j]);   // Im += Ai*Wr
      }
    async_wait0();
    __syncthreads();
  }

  const int cc = lane & 15, rb = (lane >> 4) * 8;
#pragma unroll
  for (int i = 0; i < 2; ++i)
#pragma unroll
    for (int j = 0; j < 2; ++j) {
      v8f cre = arr[i][j] - aii[i][j];
      v8f cim = ari[i][j] + air[i][j];
      const int col  = col0 + wc + 16 * j + cc;
      const int rowb = row0 + wr + 16 * i + rb;
      const float2 bb = bias[col];
#pragma unroll
      for (int v = 0; v < 8; ++v) {
        float re = cre[v] + bb.x;
        float im = cim[v] + bb.y;
        if (relu) { re = fmaxf(re, 0.f); im = fmaxf(im, 0.f); }
        const size_t r = (size_t)(rowb + v);
        if (Cr32) {
          Cr32[r * (size_t)ldc32 + col] = re;
          Ci32[r * (size_t)ldc32 + col] = im;
        }
        if (Cr16) {
          const size_t idx = trans ? ((size_t)col * (size_t)ldc16 + r)
                                   : (r * (size_t)ldc16 + col);
          Cr16[idx] = (_Float16)re;
          Ci16[idx] = (_Float16)im;
        }
      }
    }
}

// ---------------------------------------------------------------------------
// Attention: per (b, head, 64-query-row block). logits = Re(Q conj(K)^T)
// = Qr*Kr^T + Qi*Ki^T, all S=1024 logits kept in LDS (64x1028 fp32, 263KB),
// softmax in LDS, then O = P*V (complex) with WMMA off a transposed V plane.
// ---------------------------------------------------------------------------
__global__ __launch_bounds__(128) void attn_kernel(
    const _Float16* __restrict__ Qr, const _Float16* __restrict__ Qi, int ldq,
    const _Float16* __restrict__ Kr, const _Float16* __restrict__ Ki, int ldk,
    const _Float16* __restrict__ Vtr, const _Float16* __restrict__ Vti, int ldvt,
    _Float16* __restrict__ Or, _Float16* __restrict__ Oi, int ldo,
    int T, int S, float scale)
{
  __shared__ float lg[64][1028];   // 263KB of the 320KB WGP LDS

  const int b = blockIdx.z, h = blockIdx.y, tb = blockIdx.x;
  const int tid = threadIdx.x, lane = tid & 31, wave = tid >> 5;
  const int frow = lane & 15, ka = (lane >> 4) * 8, kb = (lane >> 4) * 8;
  const int HD = 64;
  const int q0 = tb * 64;

  // Q fragments (held in registers for the whole block): 2 k-steps x 2 planes
  const _Float16* qbR = Qr + ((size_t)b * T + q0 + wave * 16 + frow) * ldq + h * HD;
  const _Float16* qbI = Qi + ((size_t)b * T + q0 + wave * 16 + frow) * ldq + h * HD;
  v16h qR[2], qI[2];
  qR[0] = frag_a(qbR + ka);      qR[1] = frag_a(qbR + 32 + ka);
  qI[0] = frag_a(qbI + ka);      qI[1] = frag_a(qbI + 32 + ka);

  const v8f zero8 = {0.f,0.f,0.f,0.f,0.f,0.f,0.f,0.f};
  const int cc = lane & 15, rb = (lane >> 4) * 8;

  // ---- logits = (Qr Kr^T + Qi Ki^T) * scale  -> LDS
  for (int s0 = 0; s0 < S; s0 += 16) {
    const _Float16* kbR = Kr + ((size_t)b * S + s0 + frow) * ldk + h * HD;
    const _Float16* kbI = Ki + ((size_t)b * S + s0 + frow) * ldk + h * HD;
    v16h kR0 = frag_b(kbR + kb), kR1 = frag_b(kbR + 32 + kb);
    v16h kI0 = frag_b(kbI + kb), kI1 = frag_b(kbI + 32 + kb);
    v8f acc = zero8;
    acc = wmma_f16(qR[0], kR0, acc);
    acc = wmma_f16(qR[1], kR1, acc);
    acc = wmma_f16(qI[0], kI0, acc);
    acc = wmma_f16(qI[1], kI1, acc);
#pragma unroll
    for (int v = 0; v < 8; ++v) lg[wave * 16 + rb + v][s0 + cc] = acc[v] * scale;
  }
  __syncthreads();

  // ---- softmax over S, one row per thread (64 rows)
  if (tid < 64) {
    float* row = &lg[tid][0];
    float m = -3.4e38f;
    for (int s = 0; s < S; ++s) m = fmaxf(m, row[s]);
    float sum = 0.f;
    for (int s = 0; s < S; ++s) { float e = __expf(row[s] - m); row[s] = e; sum += e; }
    float inv = 1.f / sum;
    for (int s = 0; s < S; ++s) row[s] *= inv;
  }
  __syncthreads();

  // ---- O = P * V (complex): P from LDS (fp32 -> f16), V from transposed plane
  v8f oR[4], oI[4];
#pragma unroll
  for (int j = 0; j < 4; ++j) { oR[j] = zero8; oI[j] = zero8; }

  for (int s0 = 0; s0 < S; s0 += 32) {
    const float* pp = &lg[wave * 16 + frow][s0 + ka];
    HF16 u;
#pragma unroll
    for (int i2 = 0; i2 < 8; ++i2) {
      u.v[i2]     = (_Float16)pp[i2];
      u.v[8 + i2] = (_Float16)pp[16 + i2];
    }
    v16h p = u.v;
#pragma unroll
    for (int j = 0; j < 4; ++j) {
      const _Float16* vbR = Vtr + ((size_t)h * HD + j * 16 + frow) * ldvt + (size_t)b * S + s0 + kb;
      const _Float16* vbI = Vti + ((size_t)h * HD + j * 16 + frow) * ldvt + (size_t)b * S + s0 + kb;
      v16h vR = frag_b(vbR);
      v16h vI = frag_b(vbI);
      oR[j] = wmma_f16(p, vR, oR[j]);
      oI[j] = wmma_f16(p, vI, oI[j]);
    }
  }

#pragma unroll
  for (int j = 0; j < 4; ++j)
#pragma unroll
    for (int v = 0; v < 8; ++v) {
      const size_t row = (size_t)b * T + q0 + wave * 16 + rb + v;
      Or[row * ldo + h * HD + j * 16 + cc] = (_Float16)oR[j][v];
      Oi[row * ldo + h * HD + j * 16 + cc] = (_Float16)oI[j][v];
    }
}

// ---------------------------------------------------------------------------
// Complex LayerNorm (whitening form) over one row of D=1024, with residual add.
// ---------------------------------------------------------------------------
__device__ __forceinline__ float block_reduce_sum(float v, float* red) {
  const int tid = threadIdx.x;
  red[tid] = v; __syncthreads();
  for (int s = 128; s > 0; s >>= 1) {
    if (tid < s) red[tid] += red[tid + s];
    __syncthreads();
  }
  float r = red[0]; __syncthreads();
  return r;
}

__global__ __launch_bounds__(256) void cln_kernel(
    const float* __restrict__ resR, const float* __restrict__ resI,
    const float* __restrict__ addR, const float* __restrict__ addI,
    const float* __restrict__ w3, const float2* __restrict__ bC,
    float* __restrict__ oR, float* __restrict__ oI,
    _Float16* __restrict__ oR16, _Float16* __restrict__ oI16,
    float2* __restrict__ oC, int D)
{
  __shared__ float red[256];
  const int tid = threadIdx.x;
  const size_t base = (size_t)blockIdx.x * D;

  float xr[4], xi[4];
  float sr = 0, si = 0, srr = 0, sii = 0, sri = 0;
#pragma unroll
  for (int e = 0; e < 4; ++e) {
    const int c = tid + 256 * e;
    const float r = resR[base + c] + addR[base + c];
    const float m = resI[base + c] + addI[base + c];
    xr[e] = r; xi[e] = m;
    sr += r; si += m; srr += r * r; sii += m * m; sri += r * m;
  }
  sr  = block_reduce_sum(sr,  red);
  si  = block_reduce_sum(si,  red);
  srr = block_reduce_sum(srr, red);
  sii = block_reduce_sum(sii, red);
  sri = block_reduce_sum(sri, red);

  const float invD = 1.0f / (float)D;
  const float mur = sr * invD, mui = si * invD;
  const float vrr = srr * invD - mur * mur + EPSF;
  const float vii = sii * invD - mui * mui + EPSF;
  const float vri = sri * invD - mur * mui;
  const float s  = sqrtf(vrr * vii - vri * vri);
  const float t  = sqrtf(vrr + vii + 2.0f * s);
  const float inv = 1.0f / (s * t);
  const float rrr = (vii + s) * inv;
  const float rii = (vrr + s) * inv;
  const float rri = -vri * inv;

#pragma unroll
  for (int e = 0; e < 4; ++e) {
    const int c = tid + 256 * e;
    const float zr = xr[e] - mur, zi = xi[e] - mui;
    const float nr = rrr * zr + rri * zi;
    const float ni = rri * zr + rii * zi;
    const float wrr = w3[c * 3 + 0], wri = w3[c * 3 + 1], wii = w3[c * 3 + 2];
    const float2 bb = bC[c];
    const float ore = wrr * nr + wri * ni + bb.x;
    const float oim = wri * nr + wii * ni + bb.y;
    if (oR)   { oR[base + c] = ore; oI[base + c] = oim; }
    if (oR16) { oR16[base + c] = (_Float16)ore; oI16[base + c] = (_Float16)oim; }
    if (oC)   { oC[base + c] = make_float2(ore, oim); }
  }
}

// ---------------------------------------------------------------------------
// Conversions
// ---------------------------------------------------------------------------
__global__ void f2h_kernel(const float* __restrict__ in, _Float16* __restrict__ out, size_t n) {
  size_t i = (size_t)blockIdx.x * blockDim.x + threadIdx.x;
  const size_t stride = (size_t)gridDim.x * blockDim.x;
  for (; i < n; i += stride) out[i] = (_Float16)in[i];
}
__global__ void c2p_kernel(const float2* __restrict__ in,
                           _Float16* __restrict__ outR, _Float16* __restrict__ outI, size_t n) {
  size_t i = (size_t)blockIdx.x * blockDim.x + threadIdx.x;
  const size_t stride = (size_t)gridDim.x * blockDim.x;
  for (; i < n; i += stride) { outR[i] = (_Float16)in[i].x; outI[i] = (_Float16)in[i].y; }
}

// ===========================================================================
extern "C" void kernel_launch(void* const* d_in, const int* in_sizes, int n_in,
                              void* d_out, int out_size, void* d_ws, size_t ws_size,
                              hipStream_t stream)
{
  (void)in_sizes; (void)n_in; (void)out_size; (void)ws_size;
  const int B = 8, T = 1024, S = 1024, D = 1024, DFF = 4096;
  const int M = B * T;                    // 8192 rows

  const float*  tgt_re = (const float*)d_in[0];
  const float*  tgt_im = (const float*)d_in[1];
  const float*  mem_re = (const float*)d_in[2];
  const float*  mem_im = (const float*)d_in[3];
  const float2* sa_w_in  = (const float2*)d_in[4];
  const float2* sa_b_in  = (const float2*)d_in[5];
  const float2* sa_w_out = (const float2*)d_in[6];
  const float2* sa_b_out = (const float2*)d_in[7];
  const float2* ca_w_in  = (const float2*)d_in[8];
  const float2* ca_b_in  = (const float2*)d_in[9];
  const float2* ca_w_out = (const float2*)d_in[10];
  const float2* ca_b_out = (const float2*)d_in[11];
  const float2* w1c = (const float2*)d_in[12];
  const float2* b1c = (const float2*)d_in[13];
  const float2* w2c = (const float2*)d_in[14];
  const float2* b2c = (const float2*)d_in[15];
  const float*  ln1w = (const float*)d_in[16];
  const float2* ln1b = (const float2*)d_in[17];
  const float*  ln2w = (const float*)d_in[18];
  const float2* ln2b = (const float2*)d_in[19];
  const float*  ln3w = (const float*)d_in[20];
  const float2* ln3b = (const float2*)d_in[21];

  char* ws = (char*)d_ws;
  size_t off = 0;
  auto take = [&](size_t bytes) -> char* {
    char* p = ws + off; off += (bytes + 255) & ~(size_t)255; return p;
  };

  // weight planes (f16)
  _Float16* saWinR  = (_Float16*)take((size_t)3072 * 1024 * 2);
  _Float16* saWinI  = (_Float16*)take((size_t)3072 * 1024 * 2);
  _Float16* saWoutR = (_Float16*)take((size_t)1024 * 1024 * 2);
  _Float16* saWoutI = (_Float16*)take((size_t)1024 * 1024 * 2);
  _Float16* caWinR  = (_Float16*)take((size_t)3072 * 1024 * 2);
  _Float16* caWinI  = (_Float16*)take((size_t)3072 * 1024 * 2);
  _Float16* caWoutR = (_Float16*)take((size_t)1024 * 1024 * 2);
  _Float16* caWoutI = (_Float16*)take((size_t)1024 * 1024 * 2);
  _Float16* w1R = (_Float16*)take((size_t)DFF * 1024 * 2);
  _Float16* w1I = (_Float16*)take((size_t)DFF * 1024 * 2);
  _Float16* w2R = (_Float16*)take((size_t)1024 * DFF * 2);
  _Float16* w2I = (_Float16*)take((size_t)1024 * DFF * 2);
  // activation planes (f16)
  _Float16* tgt16R = (_Float16*)take((size_t)M * D * 2);
  _Float16* tgt16I = (_Float16*)take((size_t)M * D * 2);
  _Float16* mem16R = (_Float16*)take((size_t)M * D * 2);
  _Float16* mem16I = (_Float16*)take((size_t)M * D * 2);
  _Float16* actR   = (_Float16*)take((size_t)M * D * 2);   // x1_16 then x2_16
  _Float16* actI   = (_Float16*)take((size_t)M * D * 2);
  // contiguous region: qk planes (also aliased by FFN hidden later)
  _Float16* qkR  = (_Float16*)take((size_t)M * 2048 * 2);
  _Float16* qkI  = (_Float16*)take((size_t)M * 2048 * 2);
  _Float16* VtR  = (_Float16*)take((size_t)1024 * M * 2);  // transposed V plane
  _Float16* VtI  = (_Float16*)take((size_t)1024 * M * 2);
  _Float16* O16R = (_Float16*)take((size_t)M * D * 2);
  _Float16* O16I = (_Float16*)take((size_t)M * D * 2);
  // FFN hidden aliases qk..O16 (both phases are disjoint in time): 64MB + 64MB
  _Float16* h16R = qkR;   // spans qkR+qkI  = 8192*4096 halfs
  _Float16* h16I = VtR;   // spans VtR+VtI+O16R+O16I = 8192*4096 halfs
  // cross-attn q/k planes carved from the qk region
  _Float16* cqR = qkR;                          _Float16* cqI = qkI;
  _Float16* ckR = qkR + (size_t)M * 1024;       _Float16* ckI = qkI + (size_t)M * 1024;
  // fp32 planes
  float* gCr = (float*)take((size_t)M * D * 4);
  float* gCi = (float*)take((size_t)M * D * 4);
  float* x1R = (float*)take((size_t)M * D * 4);
  float* x1I = (float*)take((size_t)M * D * 4);
  float* x2R = (float*)take((size_t)M * D * 4);
  float* x2I = (float*)take((size_t)M * D * 4);

  const dim3 blk128(128), blk256(256);

  // ---- phase 0: precision conversion to f16 planes -----------------------
  f2h_kernel<<<2048, blk256, 0, stream>>>(tgt_re, tgt16R, (size_t)M * D);
  f2h_kernel<<<2048, blk256, 0, stream>>>(tgt_im, tgt16I, (size_t)M * D);
  f2h_kernel<<<2048, blk256, 0, stream>>>(mem_re, mem16R, (size_t)M * D);
  f2h_kernel<<<2048, blk256, 0, stream>>>(mem_im, mem16I, (size_t)M * D);
  c2p_kernel<<<2048, blk256, 0, stream>>>(sa_w_in,  saWinR,  saWinI,  (size_t)3072 * 1024);
  c2p_kernel<<<2048, blk256, 0, stream>>>(sa_w_out, saWoutR, saWoutI, (size_t)1024 * 1024);
  c2p_kernel<<<2048, blk256, 0, stream>>>(ca_w_in,  caWinR,  caWinI,  (size_t)3072 * 1024);
  c2p_kernel<<<2048, blk256, 0, stream>>>(ca_w_out, caWoutR, caWoutI, (size_t)1024 * 1024);
  c2p_kernel<<<2048, blk256, 0, stream>>>(w1c, w1R, w1I, (size_t)DFF * 1024);
  c2p_kernel<<<2048, blk256, 0, stream>>>(w2c, w2R, w2I, (size_t)1024 * DFF);

  // ---- phase 1: self-attention -------------------------------------------
  // QK projection (N=2048) -> qk planes
  cgemm_kernel<<<dim3(2048 / 64, M / 64), blk128, 0, stream>>>(
      tgt16R, tgt16I, D, saWinR, saWinI, D, sa_b_in,
      nullptr, nullptr, 0, qkR, qkI, 2048, D, 0, 0);
  // V projection (N=1024) -> transposed plane Vt[1024][8192]
  cgemm_kernel<<<dim3(1024 / 64, M / 64), blk128, 0, stream>>>(
      tgt16R, tgt16I, D, saWinR + (size_t)2048 * D, saWinI + (size_t)2048 * D, D,
      sa_b_in + 2048, nullptr, nullptr, 0, VtR, VtI, M, D, 1, 0);
  attn_kernel<<<dim3(T / 64, 16, B), blk128, 0, stream>>>(
      qkR, qkI, 2048, qkR + 1024, qkI + 1024, 2048, VtR, VtI, M,
      O16R, O16I, D, T, S, 0.125f);
  // out projection -> fp32 planes
  cgemm_kernel<<<dim3(1024 / 64, M / 64), blk128, 0, stream>>>(
      O16R, O16I, D, saWoutR, saWoutI, D, sa_b_out,
      gCr, gCi, D, nullptr, nullptr, 0, D, 0, 0);
  // x1 = cln(tgt + sa_out)
  cln_kernel<<<M, blk256, 0, stream>>>(tgt_re, tgt_im, gCr, gCi, ln1w, ln1b,
                                       x1R, x1I, actR, actI, nullptr, D);

  // ---- phase 2: cross-attention ------------------------------------------
  cgemm_kernel<<<dim3(1024 / 64, M / 64), blk128, 0, stream>>>(
      actR, actI, D, caWinR, caWinI, D, ca_b_in,
      nullptr, nullptr, 0, cqR, cqI, 1024, D, 0, 0);
  cgemm_kernel<<<dim3(1024 / 64, M / 64), blk128, 0, stream>>>(
      mem16R, mem16I, D, caWinR + (size_t)1024 * D, caWinI + (size_t)1024 * D, D,
      ca_b_in + 1024, nullptr, nullptr, 0, ckR, ckI, 1024, D, 0, 0);
  cgemm_kernel<<<dim3(1024 / 64, M / 64), blk128, 0, stream>>>(
      mem16R, mem16I, D, caWinR + (size_t)2048 * D, caWinI + (size_t)2048 * D, D,
      ca_b_in + 2048, nullptr, nullptr, 0, VtR, VtI, M, D, 1, 0);
  attn_kernel<<<dim3(T / 64, 16, B), blk128, 0, stream>>>(
      cqR, cqI, 1024, ckR, ckI, 1024, VtR, VtI, M,
      O16R, O16I, D, T, S, 0.125f);
  cgemm_kernel<<<dim3(1024 / 64, M / 64), blk128, 0, stream>>>(
      O16R, O16I, D, caWoutR, caWoutI, D, ca_b_out,
      gCr, gCi, D, nullptr, nullptr, 0, D, 0, 0);
  // x2 = cln(x1 + ca_out)
  cln_kernel<<<M, blk256, 0, stream>>>(x1R, x1I, gCr, gCi, ln2w, ln2b,
                                       x2R, x2I, actR, actI, nullptr, D);

  // ---- phase 3: FFN -------------------------------------------------------
  // h = crelu(x2 * w1^T + b1) -> f16 planes (aliased region)
  cgemm_kernel<<<dim3(DFF / 64, M / 64), blk128, 0, stream>>>(
      actR, actI, D, w1R, w1I, D, b1c,
      nullptr, nullptr, 0, h16R, h16I, DFF, D, 0, 1);
  // ff = h * w2^T + b2 -> fp32 planes
  cgemm_kernel<<<dim3(1024 / 64, M / 64), blk128, 0, stream>>>(
      h16R, h16I, DFF, w2R, w2I, DFF, b2c,
      gCr, gCi, D, nullptr, nullptr, 0, DFF, 0, 0);
  // out = cln(x2 + ff) -> interleaved complex64 into d_out
  cln_kernel<<<M, blk256, 0, stream>>>(x2R, x2I, gCr, gCi, ln3w, ln3b,
                                       nullptr, nullptr, nullptr, nullptr,
                                       (float2*)d_out, D);
}
